// GATNet_79783312490627
// MI455X (gfx1250) — matrix operand order repaired and checked by reading, not compiled, not run on hardware
//
#include <hip/hip_runtime.h>
#include <math.h>
#include <stdint.h>

// ---------------------------------------------------------------------------
// Types for WMMA fragments (CDNA5 / gfx1250, wave32)
// ---------------------------------------------------------------------------
typedef __attribute__((ext_vector_type(16))) __bf16        v16bf;
typedef __attribute__((ext_vector_type(8)))  float         v8f;
typedef __attribute__((ext_vector_type(4)))  unsigned int  u32x4;
typedef __attribute__((ext_vector_type(4)))  unsigned int  v4u;
typedef __attribute__((ext_vector_type(8)))  int           v8i;
typedef __attribute__((ext_vector_type(4)))  int           v4i;
typedef unsigned short u16b;   // raw bf16 storage

union FragBF { u32x4 u[2]; v16bf v; };

// Tensor Data Mover availability (device pass only; host pass parses fallback)
#if defined(__HIP_DEVICE_COMPILE__) && __has_builtin(__builtin_amdgcn_tensor_load_to_lds) && \
    __has_builtin(__builtin_amdgcn_s_wait_tensorcnt)
#define HAS_TDM 1
#endif

// ---------------------------------------------------------------------------
// Scalar helpers
// ---------------------------------------------------------------------------
__device__ __forceinline__ u16b f2bf(float f) {
  unsigned u = __float_as_uint(f);
  unsigned r = u + 0x7FFFu + ((u >> 16) & 1u);   // round-to-nearest-even
  return (u16b)(r >> 16);
}
__device__ __forceinline__ float bf2f(u16b b) {
  return __uint_as_float(((unsigned)b) << 16);
}
// order-preserving float -> uint key (for atomicMax-based float max)
__device__ __forceinline__ unsigned fkey(float f) {
  unsigned u = __float_as_uint(f);
  return (u & 0x80000000u) ? ~u : (u | 0x80000000u);
}
__device__ __forceinline__ float funkey(unsigned k) {
  unsigned u = (k & 0x80000000u) ? (k & 0x7FFFFFFFu) : ~k;
  return __uint_as_float(u);
}
__device__ __forceinline__ float elu_f(float x)  { return x > 0.f ? x : (expf(x) - 1.f); }
__device__ __forceinline__ float lrelu_f(float x){ return x > 0.f ? x : 0.2f * x; }

#ifdef HAS_TDM
// Issue one TDM 2-D tile load: 32 (K, contiguous) x 128 (M rows) bf16 elements
// from global (row stride = K elements) into LDS at lds_off (tight 128x32 tile).
// OOB rows (>= rows_rem) are zero-filled by the TDM.
// D# layout per CDNA5 ISA.
__device__ __forceinline__ void tdm_load_a_tile(const u16b* gbase, unsigned lds_off,
                                                unsigned rows_rem, unsigned k_rem,
                                                unsigned K) {
  unsigned long long ga = (unsigned long long)(uintptr_t)gbase;
  v4u g0;
  g0[0] = 1u;                                              // count=1 (valid), user mode
  g0[1] = lds_off;                                         // lds_addr
  g0[2] = (unsigned)(ga & 0xFFFFFFFFu);                    // global_addr[31:0]
  g0[3] = (unsigned)((ga >> 32) & 0x01FFFFFFu)             // global_addr[56:32]
        | 0x80000000u;                                     // type = 2 ("image")
  unsigned long long s0 = (unsigned long long)K;           // tensor_dim0_stride
  v8i g1;
  g1[0] = (int)(1u << 16);                                 // data_size = 1 (2 bytes)
  g1[1] = (int)((k_rem & 0xFFFFu) << 16);                  // tensor_dim0[15:0]
  g1[2] = (int)(((k_rem >> 16) & 0xFFFFu)                  // tensor_dim0[31:16]
        |       ((rows_rem & 0xFFFFu) << 16));             // tensor_dim1[15:0]
  g1[3] = (int)(((rows_rem >> 16) & 0xFFFFu)               // tensor_dim1[31:16]
        |       (32u << 16));                              // tile_dim0 = 32
  g1[4] = (int)128u;                                       // tile_dim1 = 128, tile_dim2 = 0
  g1[5] = (int)(unsigned)(s0 & 0xFFFFFFFFu);               // dim0_stride[31:0]
  g1[6] = (int)(unsigned)((s0 >> 32) & 0xFFFFu);           // dim0_stride[47:32]
  g1[7] = 0;
  v4i gz = {0, 0, 0, 0};
#if defined(__clang_major__) && (__clang_major__ >= 23)
  v8i gz8 = {0, 0, 0, 0, 0, 0, 0, 0};
  __builtin_amdgcn_tensor_load_to_lds(g0, g1, gz, gz, gz8, 0);
#else
  __builtin_amdgcn_tensor_load_to_lds(g0, g1, gz, gz, 0);
#endif
}
#endif // HAS_TDM

// ---------------------------------------------------------------------------
// Fill kernels
// ---------------------------------------------------------------------------
__global__ void fill_f32_k(float* p, float v, long n) {
  long i = (long)blockIdx.x * blockDim.x + threadIdx.x;
  if (i < n) p[i] = v;
}
__global__ void fill_u32_k(unsigned* p, unsigned v, long n) {
  long i = (long)blockIdx.x * blockDim.x + threadIdx.x;
  if (i < n) p[i] = v;
}

// f32 [Rs,Cs] -> bf16 [Rd,Cd], zero-padded
__global__ void cvt_pad_bf16_k(const float* __restrict__ src, u16b* __restrict__ dst,
                               int Rs, int Cs, int Rd, int Cd) {
  long i = (long)blockIdx.x * blockDim.x + threadIdx.x;
  long n = (long)Rd * Cd;
  if (i >= n) return;
  int r = (int)(i / Cd), c = (int)(i % Cd);
  float v = (r < Rs && c < Cs) ? src[(long)r * Cs + c] : 0.f;
  dst[i] = f2bf(v);
}

// ---------------------------------------------------------------------------
// WMMA bf16 GEMM:  C[M,N] = act(A[M,K] @ B[K,N] + bias)
//   A,B bf16 row-major, K multiple of 32, N multiple of 64.
//   Block: 256 threads (8 waves), 128x64 tile; wave = 32x32 (2x2 WMMA tiles).
//   Double-buffered LDS; A tile DMA'd via tensor_load_to_lds (TENSORcnt) and
//   overlapped with WMMA on the previous tile. ACT: 0 = none, 1 = relu
// ---------------------------------------------------------------------------
template <int ACT>
__global__ __launch_bounds__(256)
void gemm_bf16_k(const u16b* __restrict__ A, const u16b* __restrict__ Bm,
                 const float* __restrict__ bias,
                 float* __restrict__ Cf, u16b* __restrict__ Cbf,
                 int M, int N, int K) {
  __shared__ u16b As[2][128 * 32];  // [m][k]
  __shared__ u16b Bs[2][64 * 32];   // [n][k] (transposed stage -> contiguous K frags)

  const int tid  = threadIdx.x;
  const int m0   = blockIdx.x * 128;
  const int n0   = blockIdx.y * 64;
  const int w    = tid >> 5;
  const int lane = tid & 31;
  const int wm   = (w >> 1) * 32;  // 0..96
  const int wn   = (w & 1) * 32;   // 0..32
  const int half = lane >> 4;      // lane group (0: lanes 0-15, 1: lanes 16-31)
  const int lid  = lane & 15;
  const int kk   = tid >> 3;       // B stage: K row 0..31
  const int seg  = tid & 7;        // B stage: 8-elem N segment

  const v8f zero = {0.f, 0.f, 0.f, 0.f, 0.f, 0.f, 0.f, 0.f};
  v8f acc[2][2];
  acc[0][0] = zero; acc[0][1] = zero; acc[1][0] = zero; acc[1][1] = zero;

  // --- staging helpers ---
  auto load_b = [&](int k0) -> u32x4 {                      // global fetch (async till use)
    const u16b* bp = Bm + (size_t)(k0 + kk) * N + n0 + seg * 8;
    __builtin_prefetch(bp + (size_t)32 * N, 0, 1);          // global_prefetch_b8
    return *(const u32x4*)bp;
  };
  auto scatter_b = [&](u32x4 v, int buf) {                  // transpose into LDS [n][k]
    const u16b* tp = (const u16b*)&v;
#pragma unroll
    for (int j = 0; j < 8; ++j)
      Bs[buf][(seg * 8 + j) * 32 + kk] = tp[j];
  };
  auto stage_a = [&](int k0, int buf) {
#ifdef HAS_TDM
    if (tid < 32)   // wave 0 issues DMA; OOB rows >= M-m0 zero-filled by TDM
      tdm_load_a_tile(A + (size_t)m0 * K + k0, (unsigned)(uintptr_t)&As[buf][0],
                      (unsigned)(M - m0), (unsigned)(K - k0), (unsigned)K);
#else
    int r = tid >> 1, sg = tid & 1;
    int gr = m0 + r;
    u32x4 val = (u32x4)0u;
    if (gr < M) val = *(const u32x4*)(A + (size_t)gr * K + k0 + sg * 16);
    *(u32x4*)(&As[buf][0] + r * 32 + sg * 16) = val;
#endif
  };

  // --- prologue: stage tile 0 into buffer 0 ---
  const int nk = K >> 5;
  stage_a(0, 0);
  scatter_b(load_b(0), 0);
#ifdef HAS_TDM
  if (tid < 32) __builtin_amdgcn_s_wait_tensorcnt(0);
#endif
  __syncthreads();

  for (int ks = 0; ks < nk; ++ks) {
    const int cur = ks & 1, nxt = cur ^ 1;
    const bool have_next = (ks + 1) < nk;
    u32x4 bnext = (u32x4)0u;
    if (have_next) {
      stage_a((ks + 1) << 5, nxt);     // async DMA into other buffer
      bnext = load_b((ks + 1) << 5);   // global load in flight during WMMAs
    }

    // Build fragments per ISA 16-bit layouts.
    // A 16x32: lane row = lid; elems 0..7 -> K = 8*half.., elems 8..15 -> 16+8*half..
    FragBF af[2], bfr[2];
#pragma unroll
    for (int i = 0; i < 2; ++i) {
      const u16b* rp = &As[cur][0] + (wm + i * 16 + lid) * 32;
      af[i].u[0] = *(const u32x4*)(rp + 8 * half);
      af[i].u[1] = *(const u32x4*)(rp + 16 + 8 * half);
    }
    // B 32x16: lane col = lid; elems 0..7 -> K = 8*half.., elems 8..15 -> 16+8*half..
#pragma unroll
    for (int j = 0; j < 2; ++j) {
      const u16b* cp = &Bs[cur][0] + (wn + j * 16 + lid) * 32 + 8 * half;
      bfr[j].u[0] = *(const u32x4*)(cp);
      bfr[j].u[1] = *(const u32x4*)(cp + 16);
    }
#pragma unroll
    for (int i = 0; i < 2; ++i)
#pragma unroll
      for (int j = 0; j < 2; ++j)
        acc[i][j] = __builtin_amdgcn_wmma_f32_16x16x32_bf16(
            false, af[i].v, false, bfr[j].v, (short)0, acc[i][j], false, false);

    if (have_next) {
      scatter_b(bnext, nxt);           // loadcnt wait lands here, after WMMAs issued
#ifdef HAS_TDM
      if (tid < 32) __builtin_amdgcn_s_wait_tensorcnt(0);
#endif
    }
    __syncthreads();                   // publish next buffer (incl. dscnt drain)
  }

  // Store: C/D layout -> VGPR r holds (M = 8*half + r, N = lid)
#pragma unroll
  for (int i = 0; i < 2; ++i) {
#pragma unroll
    for (int j = 0; j < 2; ++j) {
      int gn = n0 + wn + j * 16 + lid;
      float bv = bias ? bias[gn] : 0.f;
#pragma unroll
      for (int r = 0; r < 8; ++r) {
        int gm = m0 + wm + i * 16 + 8 * half + r;
        if (gm < M) {
          float v = acc[i][j][r] + bv;
          if (ACT == 1) v = fmaxf(v, 0.f);
          size_t o = (size_t)gm * N + gn;
          if (Cf)  Cf[o]  = v;
          if (Cbf) Cbf[o] = f2bf(v);
        }
      }
    }
  }
}

// ---------------------------------------------------------------------------
// GAT attention kernels
// ---------------------------------------------------------------------------
// asrc[n,h] = <h[n, h*C:(h+1)*C], a_src[h]>, same for adst. grid(NP,H), block 128.
__global__ void attn_scores_k(const u16b* __restrict__ h,
                              const float* __restrict__ a_src,
                              const float* __restrict__ a_dst,
                              float* __restrict__ asrc, float* __restrict__ adst,
                              int H, int C) {
  int n = blockIdx.x, hh = blockIdx.y;
  long base = (long)n * H * C + (long)hh * C;
  __shared__ float r1[128], r2[128];
  float s1 = 0.f, s2 = 0.f;
  for (int c = threadIdx.x; c < C; c += 128) {
    float hv = bf2f(h[base + c]);
    s1 += hv * a_src[hh * C + c];
    s2 += hv * a_dst[hh * C + c];
  }
  r1[threadIdx.x] = s1; r2[threadIdx.x] = s2;
  __syncthreads();
  for (int off = 64; off > 0; off >>= 1) {
    if (threadIdx.x < off) { r1[threadIdx.x] += r1[threadIdx.x + off];
                             r2[threadIdx.x] += r2[threadIdx.x + off]; }
    __syncthreads();
  }
  if (threadIdx.x == 0) {
    asrc[(long)n * H + hh] = r1[0];
    adst[(long)n * H + hh] = r2[0];
  }
}

// e = leakyrelu(asrc[src]+adst[dst]); segment max via ordered-key atomicMax.
// Edges e<E from edge_index, e>=E are self loops. One thread per (edge, head).
__global__ void edge_logits_k(const int* __restrict__ ei,
                              const float* __restrict__ asrc,
                              const float* __restrict__ adst,
                              float* __restrict__ ebuf, unsigned* __restrict__ emax_u,
                              int E, int NP, int H) {
  long idx = (long)blockIdx.x * blockDim.x + threadIdx.x;
  long tot = (long)(E + NP) * H;
  if (idx >= tot) return;
  int e = (int)(idx / H), hh = (int)(idx % H);
  int s, d;
  if (e < E) { s = ei[e]; d = ei[E + e]; } else { s = d = e - E; }
  float v = lrelu_f(asrc[(long)s * H + hh] + adst[(long)d * H + hh]);
  ebuf[idx] = v;
  atomicMax(&emax_u[(long)d * H + hh], fkey(v));
}

// w = exp(e - emax[dst]); denom[dst] += w
__global__ void edge_expsum_k(const int* __restrict__ ei,
                              float* __restrict__ ebuf,
                              const unsigned* __restrict__ emax_u,
                              float* __restrict__ denom,
                              int E, int NP, int H) {
  long idx = (long)blockIdx.x * blockDim.x + threadIdx.x;
  long tot = (long)(E + NP) * H;
  if (idx >= tot) return;
  int e = (int)(idx / H), hh = (int)(idx % H);
  int d = (e < E) ? ei[E + e] : (e - E);
  float w = expf(ebuf[idx] - funkey(emax_u[(long)d * H + hh]));
  ebuf[idx] = w;
  atomicAdd(&denom[(long)d * H + hh], w);
}

// out[dst, f] += (w / (denom[dst]+eps)) * h[src, f].  grid(E+NP, F/256), block 256.
__global__ void edge_aggregate_k(const int* __restrict__ ei,
                                 const u16b* __restrict__ h,
                                 const float* __restrict__ wbuf,
                                 const float* __restrict__ denom,
                                 float* __restrict__ out,
                                 int E, int NP, int H, int C) {
  int e = blockIdx.x;
  int f = blockIdx.y * blockDim.x + threadIdx.x;
  int F = H * C;
  if (f >= F) return;
  int s, d;
  if (e < E) { s = ei[e]; d = ei[E + e]; } else { s = d = e - E; }
  int hh = f / C;
  float w  = wbuf[(long)e * H + hh];
  float dn = denom[(long)d * H + hh] + 1e-16f;
  atomicAdd(&out[(long)d * F + f], (w / dn) * bf2f(h[(long)s * F + f]));
}

// bf16(elu(in + bias[f]))
__global__ void bias_elu_bf16_k(const float* __restrict__ in, const float* __restrict__ bias,
                                u16b* __restrict__ out, long n, int F) {
  long i = (long)blockIdx.x * blockDim.x + threadIdx.x;
  if (i >= n) return;
  out[i] = f2bf(elu_f(in[i] + bias[i % F]));
}

// global max pool of elu(agg + bias) over batch segments (ordered-key atomicMax)
__global__ void pool_max_k(const float* __restrict__ agg, const float* __restrict__ bias,
                           const int* __restrict__ batch, unsigned* __restrict__ pool_u,
                           long n, int F) {
  long i = (long)blockIdx.x * blockDim.x + threadIdx.x;
  if (i >= n) return;
  int node = (int)(i / F), f = (int)(i % F);
  float v = elu_f(agg[i] + bias[f]);
  atomicMax(&pool_u[(long)batch[node] * F + f], fkey(v));
}

__global__ void decode_bf16_k(const unsigned* __restrict__ in, u16b* __restrict__ out, long n) {
  long i = (long)blockIdx.x * blockDim.x + threadIdx.x;
  if (i < n) out[i] = f2bf(funkey(in[i]));
}

// row-wise l2 normalize f32 [R,Cs] -> bf16 [R,Cd] zero-padded. 1 block/row, 256 thr.
__global__ void l2norm_bf16_k(const float* __restrict__ in, u16b* __restrict__ out,
                              int Cs, int Cd) {
  int r = blockIdx.x;
  __shared__ float red[256];
  float s = 0.f;
  for (int c = threadIdx.x; c < Cs; c += 256) {
    float v = in[(long)r * Cs + c];
    s += v * v;
  }
  red[threadIdx.x] = s;
  __syncthreads();
  for (int off = 128; off > 0; off >>= 1) {
    if (threadIdx.x < off) red[threadIdx.x] += red[threadIdx.x + off];
    __syncthreads();
  }
  float inv = 1.f / fmaxf(sqrtf(red[0]), 1e-12f);
  for (int c = threadIdx.x; c < Cd; c += 256) {
    float v = (c < Cs) ? in[(long)r * Cs + c] * inv : 0.f;
    out[(long)r * Cd + c] = f2bf(v);
  }
}

// xc = [v1 | v2 | c3]  (128 | 128 | 256)
__global__ void concat3_k(const float* __restrict__ v1, const float* __restrict__ v2,
                          const float* __restrict__ c3, float* __restrict__ xc, int B) {
  long i = (long)blockIdx.x * blockDim.x + threadIdx.x;
  if (i >= (long)B * 512) return;
  int r = (int)(i / 512), c = (int)(i % 512);
  float v;
  if (c < 128)       v = v1[(long)r * 128 + c];
  else if (c < 256)  v = v2[(long)r * 128 + (c - 128)];
  else               v = c3[(long)r * 256 + (c - 256)];
  xc[i] = v;
}

// out[B,2] = in[B,128] @ Wo[128,2] + bo  (tiny; one block of 256 threads)
__global__ void final_linear_k(const float* __restrict__ in, const float* __restrict__ Wo,
                               const float* __restrict__ bo, float* __restrict__ out, int B) {
  int t = threadIdx.x;
  if (t >= B * 2) return;
  int r = t / 2, j = t % 2;
  float s = bo[j];
  for (int k = 0; k < 128; ++k) s += in[r * 128 + k] * Wo[k * 2 + j];
  out[t] = s;
}

// ---------------------------------------------------------------------------
// Host orchestration
// ---------------------------------------------------------------------------
static inline unsigned cdiv(long n, int b) { return (unsigned)((n + b - 1) / b); }

extern "C" void kernel_launch(void* const* d_in, const int* in_sizes, int n_in,
                              void* d_out, int out_size, void* d_ws, size_t ws_size,
                              hipStream_t stream) {
  (void)n_in; (void)out_size; (void)ws_size;
  // ---- inputs (setup_inputs order) ----
  const float* x1   = (const float*)d_in[0];
  const int*   ei1  = (const int*)  d_in[1];
  const int*   bat1 = (const int*)  d_in[2];
  const float* x2   = (const float*)d_in[3];
  const int*   ei2  = (const int*)  d_in[4];
  const int*   bat2 = (const int*)  d_in[5];
  const float* cell = (const float*)d_in[6];
  const float* W1   = (const float*)d_in[7];
  const float* as1  = (const float*)d_in[8];
  const float* ad1  = (const float*)d_in[9];
  const float* b1   = (const float*)d_in[10];
  const float* W2   = (const float*)d_in[11];
  const float* as2  = (const float*)d_in[12];
  const float* ad2  = (const float*)d_in[13];
  const float* b2   = (const float*)d_in[14];
  const float* Wg   = (const float*)d_in[15];
  const float* bg   = (const float*)d_in[16];
  const float* Wr1  = (const float*)d_in[17];
  const float* br1  = (const float*)d_in[18];
  const float* Wr2  = (const float*)d_in[19];
  const float* br2  = (const float*)d_in[20];
  const float* Wr3  = (const float*)d_in[21];
  const float* br3  = (const float*)d_in[22];
  const float* Wf1  = (const float*)d_in[23];
  const float* bf1  = (const float*)d_in[24];
  const float* Wf2  = (const float*)d_in[25];
  const float* bf2  = (const float*)d_in[26];
  const float* Wf3  = (const float*)d_in[27];
  const float* bf3  = (const float*)d_in[28];
  const float* Wo   = (const float*)d_in[29];
  const float* bo   = (const float*)d_in[30];
  float* outp = (float*)d_out;

  const int NP = in_sizes[0] / 78;     // 6000 nodes
  const int E  = in_sizes[1] / 2;      // 36000 edges
  const int B  = in_sizes[6] / 954;    // 128 graphs
  const int ET = E + NP;               // edges + self loops
  const int F1 = 10240, H1 = 10, C1 = 1024;
  const int F2 = 512;

  // ---- workspace carve (256B aligned) ----
  size_t off = 0;
  auto carve = [&](size_t bytes) -> void* {
    void* p = (char*)d_ws + off;
    off += (bytes + 255) & ~(size_t)255;
    return p;
  };
  u16b*     h_bf   = (u16b*)    carve((size_t)NP * F1 * 2);  // layer1 feats / elu feats (bf16)
  float*    agg    = (float*)   carve((size_t)NP * F1 * 4);  // aggregation scratch (f32)
  u16b*     h2_bf  = (u16b*)    carve((size_t)NP * F2 * 2);
  u16b*     x_bf   = (u16b*)    carve((size_t)NP * 96 * 2);
  float*    asrc   = (float*)   carve((size_t)NP * H1 * 4);
  float*    adst   = (float*)   carve((size_t)NP * H1 * 4);
  unsigned* emax_u = (unsigned*)carve((size_t)NP * H1 * 4);
  float*    denom  = (float*)   carve((size_t)NP * H1 * 4);
  float*    wbuf   = (float*)   carve((size_t)ET * H1 * 4);
  unsigned* pool_u = (unsigned*)carve((size_t)B * F2 * 4);
  u16b*     g_bf   = (u16b*)    carve((size_t)B * F2 * 2);
  float*    v1f    = (float*)   carve((size_t)B * 128 * 4);
  float*    v2f    = (float*)   carve((size_t)B * 128 * 4);
  u16b*     sbf_a  = (u16b*)    carve((size_t)B * 2048 * 2);
  u16b*     sbf_b  = (u16b*)    carve((size_t)B * 2048 * 2);
  float*    c3f    = (float*)   carve((size_t)B * 256 * 4);
  float*    xcf    = (float*)   carve((size_t)B * 512 * 4);
  u16b*     xc_bf  = (u16b*)    carve((size_t)B * 512 * 2);
  float*    f3f    = (float*)   carve((size_t)B * 128 * 4);
  u16b*     W1b  = (u16b*)carve((size_t)96 * 10240 * 2);
  u16b*     W2b  = (u16b*)carve((size_t)10240 * 512 * 2);
  u16b*     Wgb  = (u16b*)carve((size_t)512 * 128 * 2);
  u16b*     Wr1b = (u16b*)carve((size_t)960 * 2048 * 2);
  u16b*     Wr2b = (u16b*)carve((size_t)2048 * 512 * 2);
  u16b*     Wr3b = (u16b*)carve((size_t)512 * 256 * 2);
  u16b*     Wf1b = (u16b*)carve((size_t)512 * 1024 * 2);
  u16b*     Wf2b = (u16b*)carve((size_t)1024 * 512 * 2);
  u16b*     Wf3b = (u16b*)carve((size_t)512 * 128 * 2);

  auto cvt = [&](const float* s, u16b* d, int Rs, int Cs, int Rd, int Cd) {
    long n = (long)Rd * Cd;
    cvt_pad_bf16_k<<<cdiv(n, 256), 256, 0, stream>>>(s, d, Rs, Cs, Rd, Cd);
  };
  auto fillf = [&](float* p, float v, long n) {
    fill_f32_k<<<cdiv(n, 256), 256, 0, stream>>>(p, v, n);
  };
  auto fillu = [&](unsigned* p, unsigned v, long n) {
    fill_u32_k<<<cdiv(n, 256), 256, 0, stream>>>(p, v, n);
  };

  // ---- weight conversions (bf16, K-padded to /32) ----
  cvt(W1, W1b, 78, 10240, 96, 10240);
  cvt(W2, W2b, 10240, 512, 10240, 512);
  cvt(Wg, Wgb, 512, 128, 512, 128);
  cvt(Wr1, Wr1b, 954, 2048, 960, 2048);
  cvt(Wr2, Wr2b, 2048, 512, 2048, 512);
  cvt(Wr3, Wr3b, 512, 256, 512, 256);
  cvt(Wf1, Wf1b, 512, 1024, 512, 1024);
  cvt(Wf2, Wf2b, 1024, 512, 1024, 512);
  cvt(Wf3, Wf3b, 512, 128, 512, 128);

#define GEMM(ACT, A_, B_, BIAS_, CF_, CBF_, M_, N_, K_)                              \
  gemm_bf16_k<ACT><<<dim3(cdiv((M_), 128), (N_) / 64), 256, 0, stream>>>(            \
      (A_), (B_), (BIAS_), (CF_), (CBF_), (M_), (N_), (K_))

  // ---- one GAT branch ----
  auto run_branch = [&](const float* x, const int* ei, const int* batch, float* vout) {
    // layer 1: h = x @ W1  (bf16 out, no bias/act yet)
    cvt(x, x_bf, NP, 78, NP, 96);
    GEMM(0, x_bf, W1b, (const float*)nullptr, (float*)nullptr, h_bf, NP, F1, 96);
    attn_scores_k<<<dim3(NP, H1), 128, 0, stream>>>(h_bf, as1, ad1, asrc, adst, H1, C1);
    fillu(emax_u, 0u, (long)NP * H1);
    fillf(denom, 0.f, (long)NP * H1);
    edge_logits_k<<<cdiv((long)ET * H1, 256), 256, 0, stream>>>(ei, asrc, adst, wbuf, emax_u, E, NP, H1);
    edge_expsum_k<<<cdiv((long)ET * H1, 256), 256, 0, stream>>>(ei, wbuf, emax_u, denom, E, NP, H1);
    fillf(agg, 0.f, (long)NP * F1);
    edge_aggregate_k<<<dim3(ET, F1 / 256), 256, 0, stream>>>(ei, h_bf, wbuf, denom, agg, E, NP, H1, C1);
    // h1 = elu(agg + b1) -> bf16 (reuse h_bf)
    bias_elu_bf16_k<<<cdiv((long)NP * F1, 256), 256, 0, stream>>>(agg, b1, h_bf, (long)NP * F1, F1);

    // layer 2: h2 = h1 @ W2 (heads=1, C=512)
    GEMM(0, h_bf, W2b, (const float*)nullptr, (float*)nullptr, h2_bf, NP, F2, F1);
    attn_scores_k<<<dim3(NP, 1), 128, 0, stream>>>(h2_bf, as2, ad2, asrc, adst, 1, F2);
    fillu(emax_u, 0u, (long)NP);
    fillf(denom, 0.f, (long)NP);
    edge_logits_k<<<cdiv((long)ET, 256), 256, 0, stream>>>(ei, asrc, adst, wbuf, emax_u, E, NP, 1);
    edge_expsum_k<<<cdiv((long)ET, 256), 256, 0, stream>>>(ei, wbuf, emax_u, denom, E, NP, 1);
    fillf(agg, 0.f, (long)NP * F2);
    edge_aggregate_k<<<dim3(ET, F2 / 256), 256, 0, stream>>>(ei, h2_bf, wbuf, denom, agg, E, NP, 1, F2);

    // global max pool of elu(agg + b2), then v = relu(g @ Wg + bg)
    fillu(pool_u, 0u, (long)B * F2);
    pool_max_k<<<cdiv((long)NP * F2, 256), 256, 0, stream>>>(agg, b2, batch, pool_u, (long)NP * F2, F2);
    decode_bf16_k<<<cdiv((long)B * F2, 256), 256, 0, stream>>>(pool_u, g_bf, (long)B * F2);
    GEMM(1, g_bf, Wgb, bg, vout, (u16b*)nullptr, B, 128, F2);
  };

  run_branch(x1, ei1, bat1, v1f);
  run_branch(x2, ei2, bat2, v2f);

  // ---- cell MLP: l2norm -> 2048 -> 512 -> 256 (relu) ----
  l2norm_bf16_k<<<B, 256, 0, stream>>>(cell, sbf_a /*reuse as cell_bf [B,960]*/, 954, 960);
  GEMM(1, sbf_a, Wr1b, br1, (float*)nullptr, sbf_b, B, 2048, 960);
  GEMM(1, sbf_b, Wr2b, br2, (float*)nullptr, sbf_a, B, 512, 2048);
  GEMM(1, sbf_a, Wr3b, br3, c3f, (u16b*)nullptr, B, 256, 512);

  // ---- fusion head ----
  concat3_k<<<cdiv((long)B * 512, 256), 256, 0, stream>>>(v1f, v2f, c3f, xcf, B);
  l2norm_bf16_k<<<B, 256, 0, stream>>>(xcf, xc_bf, 512, 512);
  GEMM(1, xc_bf, Wf1b, bf1, (float*)nullptr, sbf_a, B, 1024, 512);
  GEMM(1, sbf_a, Wf2b, bf2, (float*)nullptr, sbf_b, B, 512, 1024);
  GEMM(1, sbf_b, Wf3b, bf3, f3f, (u16b*)nullptr, B, 128, 512);
  final_linear_k<<<1, 256, 0, stream>>>(f3f, Wo, bo, outp, B);
#undef GEMM
}
